// GatedAttention_10256381903573
// MI455X (gfx1250) — compile-verified, matrix operand order checked by reading, not run on hardware
//
#include <hip/hip_runtime.h>

// ---------- types ----------
typedef __attribute__((ext_vector_type(16))) __bf16          v16bf;
typedef __attribute__((ext_vector_type(8)))  float           v8f;
typedef __attribute__((ext_vector_type(8)))  unsigned short  ushort8;
typedef __attribute__((ext_vector_type(4)))  unsigned short  ushort4v;

#define D_DIM 1024
#define S_LEN 4096
#define M_TOT 16384   // B * S

#if __has_builtin(__builtin_amdgcn_sched_barrier)
#define SCHED_BARRIER() __builtin_amdgcn_sched_barrier(0)
#else
#define SCHED_BARRIER()
#endif

// ---------- bf16 helpers ----------
static __device__ __forceinline__ unsigned short f2bf(float f) {
  unsigned int u = __builtin_bit_cast(unsigned int, f);
  u += 0x7fffu + ((u >> 16) & 1u);              // round-to-nearest-even
  return (unsigned short)(u >> 16);
}
static __device__ __forceinline__ float bf2f(unsigned short h) {
  unsigned int u = (unsigned int)h << 16;
  return __builtin_bit_cast(float, u);
}

// WMMA 16-bit A/B fragment for one lane: two contiguous 8-element (16B) runs
// at p[0..7] and p[16..23]  (caller adds the +8 element offset for lanes>=16).
struct BfPair { ushort8 lo, hi; };
static __device__ __forceinline__ v16bf load_frag(const unsigned short* p) {
  BfPair t;
  t.lo = *reinterpret_cast<const ushort8*>(p);
  t.hi = *reinterpret_cast<const ushort8*>(p + 16);
  return __builtin_bit_cast(v16bf, t);
}

static __device__ __forceinline__ v8f wmma_bf16(v16bf a, v16bf b, v8f c) {
  return __builtin_amdgcn_wmma_f32_16x16x32_bf16(false, a, false, b, (short)0, c,
                                                 false, false);
}

// ---------- pass 0a: f32 -> bf16 convert (x) ----------
__global__ __launch_bounds__(256)
void cvt_f32_to_bf16(const float* __restrict__ in, unsigned short* __restrict__ out, int n) {
  int i = (blockIdx.x * blockDim.x + threadIdx.x) * 4;
  if (i < n) {
    float4 v = *reinterpret_cast<const float4*>(in + i);
    ushort4v o;
    o[0] = f2bf(v.x); o[1] = f2bf(v.y); o[2] = f2bf(v.z); o[3] = f2bf(v.w);
    *reinterpret_cast<ushort4v*>(out + i) = o;
  }
}

// ---------- pass 0b: W [K][N] f32 -> Wt [N][K] bf16 (LDS-tiled transpose) ----------
__global__ __launch_bounds__(256)
void transpose_cvt(const float* __restrict__ w, unsigned short* __restrict__ wt) {
  __shared__ float tile[32][33];
  int n0 = blockIdx.x * 32, k0 = blockIdx.y * 32;
  int tx = threadIdx.x, ty = threadIdx.y;                 // 32 x 8
#pragma unroll
  for (int j = 0; j < 32; j += 8)
    tile[ty + j][tx] = w[(size_t)(k0 + ty + j) * D_DIM + n0 + tx];
  __syncthreads();
#pragma unroll
  for (int j = 0; j < 32; j += 8)
    wt[(size_t)(n0 + ty + j) * D_DIM + k0 + tx] = f2bf(tile[tx][ty + j]);
}

// ---------- pass 1: QKVG projection, compile-time specialized per matrix ----
// MAT: 0=Q, 1=K, 2=V (stored transposed), 3=G (sigmoid)
// grid = (N/64, M/128); block = 256 (8 waves)
template <int MAT>
__global__ __launch_bounds__(256)
void proj_wmma(const unsigned short* __restrict__ xb,
               const unsigned short* __restrict__ w,
               const float* __restrict__ bias,
               unsigned short* __restrict__ outp) {
  const int nbase = blockIdx.x * 64;
  const int mbase = blockIdx.y * 128;
  const int tid = threadIdx.x, wave = tid >> 5, lane = tid & 31;
  const int ln = lane & 15, hi = lane >> 4;
  const int koff = hi * 8;

  const unsigned short* arow =
      xb + (size_t)(mbase + wave * 16 + ln) * D_DIM + koff;
  const unsigned short* bcol0 = w + (size_t)(nbase + ln) * D_DIM + koff;

  v8f acc[4] = {};

  // prologue: fragment set for kk = 0
  v16bf a_cur = load_frag(arow);
  v16bf b_cur[4];
#pragma unroll
  for (int t = 0; t < 4; ++t)
    b_cur[t] = load_frag(bcol0 + (size_t)t * 16 * D_DIM);

#pragma unroll 4
  for (int kk = 0; kk < D_DIM - 32; kk += 32) {
    // issue next-chunk loads, then fence so WMMAs below can't pull them down
    v16bf a_nxt = load_frag(arow + kk + 32);
    v16bf b_nxt[4];
#pragma unroll
    for (int t = 0; t < 4; ++t)
      b_nxt[t] = load_frag(bcol0 + (size_t)t * 16 * D_DIM + kk + 32);
    SCHED_BARRIER();
#pragma unroll
    for (int t = 0; t < 4; ++t) acc[t] = wmma_bf16(a_cur, b_cur[t], acc[t]);
    SCHED_BARRIER();
    a_cur = a_nxt;
#pragma unroll
    for (int t = 0; t < 4; ++t) b_cur[t] = b_nxt[t];
  }
#pragma unroll
  for (int t = 0; t < 4; ++t) acc[t] = wmma_bf16(a_cur, b_cur[t], acc[t]);

#pragma unroll
  for (int t = 0; t < 4; ++t) {
    int col = nbase + t * 16 + ln;
    float bsc = bias[col];
#pragma unroll
    for (int r = 0; r < 8; ++r) {
      int row = mbase + wave * 16 + r + 8 * hi;
      float v = acc[t][r] + bsc;
      if (MAT == 2)      outp[(size_t)col * M_TOT + row] = f2bf(v);
      else if (MAT == 3) outp[(size_t)row * D_DIM + col] =
                             f2bf(1.0f / (1.0f + __expf(-v)));
      else               outp[(size_t)row * D_DIM + col] = f2bf(v);
    }
  }
}

// ---------- pass 2: flash attention + gate + residual ----------
// grid = (S/16 query tiles, B); block = 256 (8 waves).
__global__ __launch_bounds__(256)
void attn_wmma(const unsigned short* __restrict__ qb,
               const unsigned short* __restrict__ kbuf,
               const unsigned short* __restrict__ vt,
               const unsigned short* __restrict__ gb,
               const float* __restrict__ x,
               float* __restrict__ out) {
  const int qtile = blockIdx.x, b = blockIdx.y;
  const int tid = threadIdx.x, wave = tid >> 5, lane = tid & 31;
  const int ln = lane & 15, hi = lane >> 4;
  const int koff = hi * 8;
  const int qrow0 = qtile * 16;

  __shared__ float          s_scores[16][132];   // padded (scalar access)
  __shared__ unsigned short s_p[16][128];        // bf16 probabilities
  __shared__ float          s_red[16][17];       // partial reductions
  __shared__ float          s_m[16], s_l[16], s_corr[16];

  if (tid < 16) { s_m[tid] = -3.0e38f; s_l[tid] = 0.0f; }
  __syncthreads();

  const unsigned short* qp =
      qb + (size_t)(b * S_LEN + qrow0 + ln) * D_DIM + koff;
  const int colbase = wave * 128;

  v8f acc[8] = {};

  for (int k0 = 0; k0 < S_LEN; k0 += 128) {
    // hint: start pulling next key block toward L0/L2
    if (k0 + 128 < S_LEN) {
      __builtin_prefetch(
          kbuf + (size_t)(b * S_LEN + k0 + 128 + wave * 16 + ln) * D_DIM, 0, 3);
      __builtin_prefetch(
          vt + (size_t)(colbase + ln) * M_TOT + (b * S_LEN + k0 + 128), 0, 3);
    }

    // --- scores: depth-2 pipelined Q.K^T (one WMMA per chunk) ---
    const unsigned short* kp =
        kbuf + (size_t)(b * S_LEN + k0 + wave * 16 + ln) * D_DIM + koff;
    v8f sc = {};
    v16bf qa0 = load_frag(qp);
    v16bf ka0 = load_frag(kp);
    v16bf qa1 = load_frag(qp + 32);
    v16bf ka1 = load_frag(kp + 32);
#pragma unroll 4
    for (int kk = 0; kk < D_DIM - 64; kk += 32) {
      v16bf qn = load_frag(qp + kk + 64);
      v16bf kn = load_frag(kp + kk + 64);
      SCHED_BARRIER();
      sc = wmma_bf16(qa0, ka0, sc);
      SCHED_BARRIER();
      qa0 = qa1; ka0 = ka1;
      qa1 = qn;  ka1 = kn;
    }
    sc = wmma_bf16(qa0, ka0, sc);
    sc = wmma_bf16(qa1, ka1, sc);

#pragma unroll
    for (int r = 0; r < 8; ++r)
      s_scores[r + 8 * hi][wave * 16 + ln] = sc[r] * 0.03125f;  // 1/sqrt(1024)
    __syncthreads();

    // --- parallel online softmax ---
    if (tid < 128) {                       // (1) segmented row max
      int row = tid & 15, seg = tid >> 4;
      float m = -3.0e38f;
#pragma unroll
      for (int j = 0; j < 16; ++j)
        m = fmaxf(m, s_scores[row][seg * 16 + j]);
      s_red[row][seg] = m;
    }
    __syncthreads();
    if (tid < 16) {
      float mold = s_m[tid], mnew = mold;
#pragma unroll
      for (int sg = 0; sg < 8; ++sg) mnew = fmaxf(mnew, s_red[tid][sg]);
      s_corr[tid] = __expf(mold - mnew);
      s_m[tid]    = mnew;
    }
    __syncthreads();
    {                                      // (2) exp + bf16 P + partial sums
      int row = tid >> 4, seg = tid & 15;
      float mnew = s_m[row];
      float ps = 0.0f;
#pragma unroll
      for (int j = 0; j < 8; ++j) {
        float p = __expf(s_scores[row][seg * 8 + j] - mnew);
        ps += p;
        s_p[row][seg * 8 + j] = f2bf(p);
      }
      s_red[row][seg] = ps;
    }
    __syncthreads();
    if (tid < 16) {
      float ls = 0.0f;
#pragma unroll
      for (int sg = 0; sg < 16; ++sg) ls += s_red[tid][sg];
      s_l[tid] = s_l[tid] * s_corr[tid] + ls;
    }

    // --- rescale existing accumulators (per output row) ---
    float f[8];
#pragma unroll
    for (int r = 0; r < 8; ++r) f[r] = s_corr[r + 8 * hi];
#pragma unroll
    for (int t = 0; t < 8; ++t)
#pragma unroll
      for (int r = 0; r < 8; ++r) acc[t][r] *= f[r];

    // --- P @ V : A-fragments hoisted from LDS, V-fragments depth-2 pipeline --
    const unsigned short* prow = &s_p[ln][koff];
    v16bf pa[4];
#pragma unroll
    for (int kc = 0; kc < 4; ++kc) pa[kc] = load_frag(prow + kc * 32);

    const unsigned short* vb0 =
        vt + (size_t)(colbase + ln) * M_TOT + (size_t)(b * S_LEN + k0) + koff;
    auto vaddr = [&](int s) {
      return vb0 + (size_t)(s & 7) * (16 * (size_t)M_TOT) + (s >> 3) * 32;
    };
    v16bf vf0 = load_frag(vaddr(0));
    v16bf vf1 = load_frag(vaddr(1));
#pragma unroll
    for (int s = 0; s < 32; ++s) {
      v16bf vn;
      if (s < 30) vn = load_frag(vaddr(s + 2));
      SCHED_BARRIER();
      acc[s & 7] = wmma_bf16(pa[s >> 3], vf0, acc[s & 7]);
      SCHED_BARRIER();
      vf0 = vf1;
      if (s < 30) vf1 = vn;
    }
    __syncthreads();   // protect s_scores/s_p/s_l before next block
  }

  // --- epilogue: normalize, gate, residual ---
  float invl[8];
#pragma unroll
  for (int r = 0; r < 8; ++r) invl[r] = 1.0f / s_l[r + 8 * hi];
#pragma unroll
  for (int t = 0; t < 8; ++t) {
    int col = colbase + t * 16 + ln;
#pragma unroll
    for (int r = 0; r < 8; ++r) {
      size_t row = (size_t)(b * S_LEN + qrow0 + r + 8 * hi);
      float g = bf2f(gb[row * D_DIM + col]);
      float o = acc[t][r] * invl[r];
      out[row * D_DIM + col] = g * o + x[row * D_DIM + col];
    }
  }
}

// ---------- host launch ----------
extern "C" void kernel_launch(void* const* d_in, const int* in_sizes, int n_in,
                              void* d_out, int out_size, void* d_ws, size_t ws_size,
                              hipStream_t stream) {
  (void)in_sizes; (void)n_in; (void)out_size; (void)ws_size;
  const float* x  = (const float*)d_in[0];
  const float* Wq = (const float*)d_in[1];
  const float* bq = (const float*)d_in[2];
  const float* Wk = (const float*)d_in[3];
  const float* bk = (const float*)d_in[4];
  const float* Wv = (const float*)d_in[5];
  const float* bv = (const float*)d_in[6];
  const float* Wg = (const float*)d_in[7];
  const float* bg = (const float*)d_in[8];
  float* out = (float*)d_out;

  // workspace layout (bytes): xb 32MB | wt 8MB | qb 32MB | kb 32MB | vt 32MB | gb 32MB
  char* ws = (char*)d_ws;
  unsigned short* xb = (unsigned short*)(ws);
  unsigned short* wt = (unsigned short*)(ws + 33554432);
  unsigned short* qb = (unsigned short*)(ws + 41943040);
  unsigned short* kb = (unsigned short*)(ws + 75497472);
  unsigned short* vt = (unsigned short*)(ws + 109051904);
  unsigned short* gb = (unsigned short*)(ws + 142606336);

  cvt_f32_to_bf16<<<16384, 256, 0, stream>>>(x, xb, M_TOT * D_DIM);

  dim3 tb(32, 8);
  transpose_cvt<<<dim3(32, 32), tb, 0, stream>>>(Wq, wt + 0 * 1048576);
  transpose_cvt<<<dim3(32, 32), tb, 0, stream>>>(Wk, wt + 1 * 1048576);
  transpose_cvt<<<dim3(32, 32), tb, 0, stream>>>(Wv, wt + 2 * 1048576);
  transpose_cvt<<<dim3(32, 32), tb, 0, stream>>>(Wg, wt + 3 * 1048576);

  dim3 pg(16, 128);
  proj_wmma<0><<<pg, 256, 0, stream>>>(xb, wt + 0 * 1048576, bq, qb);
  proj_wmma<1><<<pg, 256, 0, stream>>>(xb, wt + 1 * 1048576, bk, kb);
  proj_wmma<2><<<pg, 256, 0, stream>>>(xb, wt + 2 * 1048576, bv, vt);
  proj_wmma<3><<<pg, 256, 0, stream>>>(xb, wt + 3 * 1048576, bg, gb);

  attn_wmma<<<dim3(256, 4), 256, 0, stream>>>(qb, kb, vt, gb, x, out);
}
